// Experts_Cute_54580444398293
// MI455X (gfx1250) — matrix operand order repaired and checked by reading, not compiled, not run on hardware
//
#include <hip/hip_runtime.h>

typedef __attribute__((ext_vector_type(16))) _Float16 v16h;
typedef __attribute__((ext_vector_type(8)))  _Float16 v8h;
typedef __attribute__((ext_vector_type(4)))  _Float16 v4h;
typedef __attribute__((ext_vector_type(8)))  float    v8f;
typedef __attribute__((ext_vector_type(4)))  float    v4f;

#define M_TILE 128
#define N_TILE 128
#define K_TILE 32
#define LDT    40   // padded LDS row (halves): 80B stride -> conflict-free b128

__global__ __launch_bounds__(256)
void experts_grouped_gemm_f16wmma(const float* __restrict__ x,
                                  const int*   __restrict__ freq,
                                  const float* __restrict__ weight,
                                  const float* __restrict__ bias,
                                  float* __restrict__ out,
                                  int E, int D_IN, int D_OUT)
{
    // Double-buffered staging tiles (f16): 2 * (10KB + 10KB) = 40KB.
    __shared__ _Float16 As[2][M_TILE][LDT];
    __shared__ _Float16 Bs[2][N_TILE][LDT];

    // Map linear row-tile id (grid.y) -> (expert e, tile within expert group).
    int tile = blockIdx.y;
    int e = 0, cnt = 0, start = 0;
    for (; e < E; ++e) {
        cnt = freq[e];
        int nt = (cnt + M_TILE - 1) / M_TILE;
        if (tile < nt) break;
        tile -= nt;
        start += cnt;
    }
    if (e >= E) return;   // uniform per block

    const int rowBase = start + tile * M_TILE;
    const int grpEnd  = start + cnt;
    const int colBase = blockIdx.x * N_TILE;   // fast grid axis: x-tile reuse in L2

    const float* __restrict__ wg = weight + (size_t)e * D_OUT * D_IN;

    const int tid  = threadIdx.x;
    const int lane = tid & 31;
    const int wave = tid >> 5;      // 8 waves: wave w owns rows [w*16, w*16+16)
    const int half = lane >> 4;
    const int mrow = lane & 15;

    // Per-thread staging slots: 4 float4 for A + 4 for B per K-stage.
    // slot s = i*256 + tid; tile-row = s>>3; col = (s&7)*4 (32 floats per row)
    const float* aPtr[4];
    const float* bPtr[4];
    int ldRow[4], ldCol[4];
#pragma unroll
    for (int i = 0; i < 4; ++i) {
        int s = i * 256 + tid;
        int r = s >> 3;
        ldRow[i] = r;
        ldCol[i] = (s & 7) * 4;
        int gr = rowBase + r;
        if (gr >= grpEnd) gr = grpEnd - 1;   // clamp OOB rows; masked at store
        aPtr[i] = x  + (size_t)gr * D_IN + ldCol[i];
        bPtr[i] = wg + (size_t)(colBase + r) * D_IN + ldCol[i];
    }

    v8f acc[8] = {};
    const int K_TILES = D_IN / K_TILE;

    // ---- Prologue: stage K-tile 0 into buffer 0 ----
    v4f aReg[4], bReg[4];
#pragma unroll
    for (int i = 0; i < 4; ++i) {
        aReg[i] = *(const v4f*)(aPtr[i]);
        bReg[i] = *(const v4f*)(bPtr[i]);
    }
#pragma unroll
    for (int i = 0; i < 4; ++i) {
        v4f a = aReg[i], b = bReg[i];
        v4h ah, bh;
        ah.x = (_Float16)a.x; ah.y = (_Float16)a.y;
        ah.z = (_Float16)a.z; ah.w = (_Float16)a.w;
        bh.x = (_Float16)b.x; bh.y = (_Float16)b.y;
        bh.z = (_Float16)b.z; bh.w = (_Float16)b.w;
        *(v4h*)&As[0][ldRow[i]][ldCol[i]] = ah;
        *(v4h*)&Bs[0][ldRow[i]][ldCol[i]] = bh;
    }
    __syncthreads();

    for (int kt = 0; kt < K_TILES; ++kt) {
        const int cur = kt & 1;
        const int nxt = cur ^ 1;
        const bool more = (kt + 1 < K_TILES);

        // Issue next stage's global loads early (overlap with WMMA chain).
        if (more) {
            const int k0 = (kt + 1) * K_TILE;
#pragma unroll
            for (int i = 0; i < 4; ++i) {
                aReg[i] = *(const v4f*)(aPtr[i] + k0);
                bReg[i] = *(const v4f*)(bPtr[i] + k0);
            }
        }

        // ---- Compute on buffer `cur` ----
        // A fragment (16x32): row mrow of wave strip; K chunks half*8, half*8+16.
        const int am = wave * 16 + mrow;
        v8h alo = *(const v8h*)&As[cur][am][half * 8];
        v8h ahi = *(const v8h*)&As[cur][am][half * 8 + 16];
        v16h afrag = __builtin_shufflevector(alo, ahi,
                        0,1,2,3,4,5,6,7,8,9,10,11,12,13,14,15);

#pragma unroll
        for (int g = 0; g < 2; ++g) {
            // Preload 4 B fragments, then 4 back-to-back WMMAs.
            v16h bfrag[4];
#pragma unroll
            for (int j = 0; j < 4; ++j) {
                const int bn = (g * 4 + j) * 16 + mrow;
                v8h blo = *(const v8h*)&Bs[cur][bn][half * 8];
                v8h bhi = *(const v8h*)&Bs[cur][bn][half * 8 + 16];
                bfrag[j] = __builtin_shufflevector(blo, bhi,
                              0,1,2,3,4,5,6,7,8,9,10,11,12,13,14,15);
            }
#pragma unroll
            for (int j = 0; j < 4; ++j) {
                acc[g * 4 + j] = __builtin_amdgcn_wmma_f32_16x16x32_f16(
                    false, afrag, false, bfrag[j],
                    (short)0, acc[g * 4 + j], false, false);
            }
        }

        // ---- Stage next tile into buffer `nxt` (other waves may still be
        // computing on `cur`; `nxt` reads were fenced by last barrier) ----
        if (more) {
#pragma unroll
            for (int i = 0; i < 4; ++i) {
                v4f a = aReg[i], b = bReg[i];
                v4h ah, bh;
                ah.x = (_Float16)a.x; ah.y = (_Float16)a.y;
                ah.z = (_Float16)a.z; ah.w = (_Float16)a.w;
                bh.x = (_Float16)b.x; bh.y = (_Float16)b.y;
                bh.z = (_Float16)b.z; bh.w = (_Float16)b.w;
                *(v4h*)&As[nxt][ldRow[i]][ldCol[i]] = ah;
                *(v4h*)&Bs[nxt][ldRow[i]][ldCol[i]] = bh;
            }
        }
        __syncthreads();   // single barrier per K-stage
    }

    // ---- Epilogue: bias + masked store ----
    // C/D layout: VGPR r -> M = r + 8*half, N = mrow (within 16x16 tile).
    const int rBaseWave = rowBase + wave * 16 + half * 8;
#pragma unroll
    for (int n = 0; n < 8; ++n) {
        const int col = colBase + n * 16 + mrow;
        const float bv = bias[(size_t)e * D_OUT + col];
#pragma unroll
        for (int r = 0; r < 8; ++r) {
            const int row = rBaseWave + r;
            if (row < grpEnd)
                out[(size_t)row * D_OUT + col] = acc[n][r] + bv;
        }
    }
}

extern "C" void kernel_launch(void* const* d_in, const int* in_sizes, int n_in,
                              void* d_out, int out_size, void* d_ws, size_t ws_size,
                              hipStream_t stream) {
    const float* x    = (const float*)d_in[0];
    const int*   freq = (const int*)  d_in[1];
    const float* w    = (const float*)d_in[2];
    const float* b    = (const float*)d_in[3];
    float* out = (float*)d_out;

    const int E     = in_sizes[1];
    const int D_OUT = in_sizes[3] / E;
    const int D_IN  = in_sizes[2] / (E * D_OUT);
    const int T     = in_sizes[0] / D_IN;

    // grid.x = column tiles (fast axis -> L2 reuse of x row tiles),
    // grid.y = worst-case row tiles across experts.
    dim3 grid(D_OUT / N_TILE, (T + M_TILE - 1) / M_TILE + E, 1);
    dim3 block(256, 1, 1);
    experts_grouped_gemm_f16wmma<<<grid, block, 0, stream>>>(
        x, freq, w, b, out, E, D_IN, D_OUT);
}
